// GroupedQueryAttention_42356967473814
// MI455X (gfx1250) — compile-verified, matrix operand order checked by reading, not run on hardware
//
#include <hip/hip_runtime.h>

// ---------------------------------------------------------------------------
// GQA forward for MI455X (gfx1250, wave32, WMMA).
// fp32 reference -> bf16 compute / fp32 accumulate via v_wmma_f32_16x16x32_bf16.
// A/K tile staging uses GLOBAL_LOAD_ASYNC_TO_LDS (ASYNCcnt) when the
// toolchain declares the builtins; otherwise falls back to VGPR staging.
// ---------------------------------------------------------------------------

typedef __attribute__((ext_vector_type(16))) __bf16 v16bf;
typedef __attribute__((ext_vector_type(8)))  __bf16 v8bf;
typedef __attribute__((ext_vector_type(8)))  float  v8f;
typedef __attribute__((ext_vector_type(4)))  int    v4i;

#define EDIM   2048
#define HEADS  16
#define GROUPS 4
#define HD     128                 // head dim
#define BATCH  2
#define SEQ    2048
#define MROWS  (BATCH * SEQ)       // 4096
#define KVCOLS (2 * EDIM / GROUPS) // 1024

#if defined(__gfx1250__) && __has_builtin(__builtin_amdgcn_global_load_async_to_lds_b128) && __has_builtin(__builtin_amdgcn_s_wait_asynccnt)
#define USE_ASYNC_LDS 1
#else
#define USE_ASYNC_LDS 0
#endif

#if USE_ASYNC_LDS
// Builtin signature (from toolchain diagnostics): (v4i addrspace(1)*,
// v4i addrspace(3)*, imm offset, imm cpol)
typedef __attribute__((address_space(1))) v4i* gv4ip;
typedef __attribute__((address_space(3))) v4i* lv4ip;
// Async copy of 16 bytes (8 bf16) global -> LDS, no VGPR staging (ASYNCcnt).
__device__ inline void async_cp16(void* lds_dst, const void* gsrc) {
  __builtin_amdgcn_global_load_async_to_lds_b128(
      (gv4ip)(void*)gsrc, (lv4ip)lds_dst, 0, 0);
}
__device__ inline void async_wait0() { __builtin_amdgcn_s_wait_asynccnt(0); }
#else
__device__ inline void async_cp16(void* lds_dst, const void* gsrc) {
  *(v8bf*)lds_dst = *(const v8bf*)gsrc;
}
__device__ inline void async_wait0() {}
#endif

// Build a v16bf A/B fragment from two non-contiguous 8-element chunks.
// 16-bit operand layout (ISA 7.12.2): lane<16 -> K {b..b+7, b+16..b+23} with
// b=0; lane>=16 -> b=8. Each chunk is a 16-byte contiguous load.
__device__ inline v16bf frag16(const __bf16* p0, const __bf16* p1) {
  v8bf lo = *(const v8bf*)p0;
  v8bf hi = *(const v8bf*)p1;
  v16bf r;
#pragma unroll
  for (int i = 0; i < 8; ++i) { r[i] = lo[i]; r[8 + i] = hi[i]; }
  return r;
}

// ---------------------------------------------------------------------------
// fp32 -> bf16 conversion
// ---------------------------------------------------------------------------
__global__ void cvt_f32_bf16(const float* __restrict__ in,
                             __bf16* __restrict__ out, int n) {
  int i = blockIdx.x * blockDim.x + threadIdx.x;
  int stride = gridDim.x * blockDim.x;
  for (; i < n; i += stride) out[i] = (__bf16)in[i];
}

// ---------------------------------------------------------------------------
// bf16 GEMM, fp32 accumulate: C = A(MxK) * B(KxN) + bias
// Block tile 128x64, BK=32, 256 threads = 8 waves, wave tile 32x32.
// Writes fp32 (Cf) and/or bf16 (Cb).
// ---------------------------------------------------------------------------
__launch_bounds__(256)
__global__ void gemm_bf16_wmma(const __bf16* __restrict__ A,
                               const __bf16* __restrict__ B,
                               const float* __restrict__ bias,
                               float* __restrict__ Cf,
                               __bf16* __restrict__ Cb,
                               int M, int N, int K) {
  __shared__ __bf16 As[128][32];   // row-major
  __shared__ __bf16 Bs[64][32];    // transposed: Bs[n][k]

  const int t    = threadIdx.x;
  const int w    = t >> 5;
  const int lane = t & 31;
  const int wm   = w & 3;          // 4 row groups of 32
  const int wn   = w >> 2;         // 2 col groups of 32
  const int m0   = blockIdx.y * 128;
  const int n0   = blockIdx.x * 64;
  const int lm   = lane & 15;
  const int kb0  = (lane < 16) ? 0 : 8;
  const int rgrp = (lane >> 4) * 8;

  v8f acc[2][2] = {};

  for (int k0 = 0; k0 < K; k0 += 32) {
    __syncthreads();
    // A tile: 128x32, 16 bf16 per thread, two async 16B copies (no VGPRs)
    {
      int row = t >> 1;
      int kb  = (t & 1) * 16;
      const __bf16* src = A + (size_t)(m0 + row) * K + k0 + kb;
      async_cp16(&As[row][kb],     src);
      async_cp16(&As[row][kb + 8], src + 8);
    }
    // B tile: 32x64 read coalesced, written transposed into Bs[n][k]
    {
      int k  = t >> 3;
      int nb = (t & 7) * 8;
      const __bf16* src = B + (size_t)(k0 + k) * N + n0 + nb;
#pragma unroll
      for (int i = 0; i < 8; ++i) Bs[nb + i][k] = src[i];
    }
    if (k0 + 32 < K) {  // gfx1250 global_prefetch_b8 for next K tile
      __builtin_prefetch(A + (size_t)(m0 + (t >> 1)) * K + k0 + 32, 0, 3);
      __builtin_prefetch(B + (size_t)(k0 + 32 + (t >> 3)) * N + n0, 0, 3);
    }
    async_wait0();      // own-wave async copies complete
    __syncthreads();    // all waves' tiles visible

    v16bf af[2], bfr[2];
#pragma unroll
    for (int mt = 0; mt < 2; ++mt) {
      int r = wm * 32 + mt * 16 + lm;
      af[mt] = frag16(&As[r][kb0], &As[r][kb0 + 16]);
    }
#pragma unroll
    for (int nt = 0; nt < 2; ++nt) {
      int c = wn * 32 + nt * 16 + lm;
      bfr[nt] = frag16(&Bs[c][kb0], &Bs[c][kb0 + 16]);
    }
#pragma unroll
    for (int mt = 0; mt < 2; ++mt)
#pragma unroll
      for (int nt = 0; nt < 2; ++nt)
        acc[mt][nt] = __builtin_amdgcn_wmma_f32_16x16x32_bf16(
            false, af[mt], false, bfr[nt], (short)0, acc[mt][nt], false, false);
  }

#pragma unroll
  for (int mt = 0; mt < 2; ++mt) {
#pragma unroll
    for (int nt = 0; nt < 2; ++nt) {
      int col  = n0 + wn * 32 + nt * 16 + lm;
      float bv = bias ? bias[col] : 0.0f;
#pragma unroll
      for (int v = 0; v < 8; ++v) {
        int row   = m0 + wm * 32 + mt * 16 + rgrp + v;
        float val = acc[mt][nt][v] + bv;
        if (Cf) Cf[(size_t)row * N + col] = val;
        if (Cb) Cb[(size_t)row * N + col] = (__bf16)val;
      }
    }
  }
}

// ---------------------------------------------------------------------------
// Flash attention (GQA). Block = (qblock of 128 rows, head, batch).
// 8 waves; each wave owns a 16-row q tile with a 16x128 fp32 accumulator.
// K/V tiles of 32 keys staged in LDS per iteration (K via async copies).
// ---------------------------------------------------------------------------
__launch_bounds__(256)
__global__ void gqa_attention_wmma(const __bf16* __restrict__ Q,
                                   const __bf16* __restrict__ KV,
                                   __bf16* __restrict__ O) {
  __shared__ __bf16 Kt[32][128];        // [key][d]
  __shared__ __bf16 Vt[128][32];        // [d][key] (transposed)
  __shared__ float  Sc[8][16][32];      // per-wave score scratch
  __shared__ __bf16 Pb[8][16][32];      // per-wave P (bf16)
  __shared__ float  Mb[8][16], Lb[8][16], Alb[8][16];

  const int t    = threadIdx.x;
  const int w    = t >> 5;
  const int lane = t & 31;
  const int b    = blockIdx.z;
  const int h    = blockIdx.y;
  const int g    = h / (HEADS / GROUPS);
  const int q0   = blockIdx.x * 128 + w * 16;
  const int lm   = lane & 15;
  const int kb0  = (lane < 16) ? 0 : 8;
  const int rgrp = (lane >> 4) * 8;
  const float scale = 0.08838834764831845f;  // 1/sqrt(128)

  // Q fragments: 16 rows x 128 dims -> 4 K=32 chunks (persistent in VGPRs)
  v16bf qf[4];
  {
    const __bf16* qrow = Q + (size_t)(b * SEQ + q0 + lm) * EDIM + h * HD;
#pragma unroll
    for (int kc = 0; kc < 4; ++kc)
      qf[kc] = frag16(qrow + kc * 32 + kb0, qrow + kc * 32 + kb0 + 16);
  }

  v8f oacc[8] = {};
  if (lane < 16) { Mb[w][lane] = -1e30f; Lb[w][lane] = 0.0f; }

  for (int kblk = 0; kblk < SEQ; kblk += 32) {
    __syncthreads();
    // Cooperative load: K tile natural (async), V tile transposed (VGPR)
    {
      int key = t >> 3;         // 0..31
      int db  = (t & 7) * 16;   // 0..112
      const __bf16* kvrow =
          KV + (size_t)(b * SEQ + kblk + key) * KVCOLS + g * (2 * HD);
      async_cp16(&Kt[key][db],     kvrow + db);
      async_cp16(&Kt[key][db + 8], kvrow + db + 8);
#pragma unroll
      for (int i = 0; i < 16; ++i)
        Vt[db + i][key] = kvrow[HD + db + i];
    }
    async_wait0();
    __syncthreads();

    // Scores: S(16x32) = Q(16x128) * K^T -- 8 WMMAs
#pragma unroll
    for (int nt = 0; nt < 2; ++nt) {
      v8f s = {};
      int key = nt * 16 + lm;
#pragma unroll
      for (int kc = 0; kc < 4; ++kc) {
        v16bf kf = frag16(&Kt[key][kc * 32 + kb0], &Kt[key][kc * 32 + kb0 + 16]);
        s = __builtin_amdgcn_wmma_f32_16x16x32_bf16(
            false, qf[kc], false, kf, (short)0, s, false, false);
      }
#pragma unroll
      for (int v = 0; v < 8; ++v) Sc[w][rgrp + v][nt * 16 + lm] = s[v];
    }

    // Online softmax: lanes 0..15 each own one q row (same-wave LDS is
    // in-order, so no barrier needed for the wave-private buffers).
    if (lane < 16) {
      float mold = Mb[w][lane];
      float mb   = mold;
#pragma unroll
      for (int j = 0; j < 32; ++j)
        mb = fmaxf(mb, Sc[w][lane][j] * scale);
      float alpha = __expf(mold - mb);
      float lsum  = 0.0f;
#pragma unroll
      for (int j = 0; j < 32; ++j) {
        float p = __expf(Sc[w][lane][j] * scale - mb);
        Pb[w][lane][j] = (__bf16)p;
        lsum += p;
      }
      Mb[w][lane]  = mb;
      Lb[w][lane]  = Lb[w][lane] * alpha + lsum;
      Alb[w][lane] = alpha;
    }
    // (reconverged: EXEC all-1s before WMMA)

    // Rescale running output, then O += P(16x32) * V(32x128) -- 8 WMMAs
    float al[8];
#pragma unroll
    for (int v = 0; v < 8; ++v) al[v] = Alb[w][rgrp + v];
#pragma unroll
    for (int nt = 0; nt < 8; ++nt)
#pragma unroll
      for (int v = 0; v < 8; ++v) oacc[nt][v] *= al[v];

    v16bf pf = frag16(&Pb[w][lm][kb0], &Pb[w][lm][kb0 + 16]);
#pragma unroll
    for (int nt = 0; nt < 8; ++nt) {
      int d = nt * 16 + lm;
      v16bf vf = frag16(&Vt[d][kb0], &Vt[d][kb0 + 16]);
      oacc[nt] = __builtin_amdgcn_wmma_f32_16x16x32_bf16(
          false, pf, false, vf, (short)0, oacc[nt], false, false);
    }
  }

  // Normalize and store attention output (bf16, [b, s, h*HD + d] layout)
  float rl[8];
#pragma unroll
  for (int v = 0; v < 8; ++v) rl[v] = 1.0f / Lb[w][rgrp + v];
#pragma unroll
  for (int nt = 0; nt < 8; ++nt) {
#pragma unroll
    for (int v = 0; v < 8; ++v) {
      int row = q0 + rgrp + v;
      int col = h * HD + nt * 16 + lm;
      O[(size_t)(b * SEQ + row) * EDIM + col] = (__bf16)(oacc[nt][v] * rl[v]);
    }
  }
}

// ---------------------------------------------------------------------------
// Launch
// ---------------------------------------------------------------------------
extern "C" void kernel_launch(void* const* d_in, const int* in_sizes, int n_in,
                              void* d_out, int out_size, void* d_ws,
                              size_t ws_size, hipStream_t stream) {
  (void)in_sizes; (void)n_in; (void)out_size; (void)ws_size;

  const float* x   = (const float*)d_in[0];
  const float* Wq  = (const float*)d_in[1];
  const float* bq  = (const float*)d_in[2];
  const float* Wkv = (const float*)d_in[3];
  const float* bkv = (const float*)d_in[4];
  const float* Wo  = (const float*)d_in[5];
  const float* bo  = (const float*)d_in[6];
  float* out = (float*)d_out;

  char* ws = (char*)d_ws;
  size_t off = 0;
  auto carve = [&](size_t bytes) -> void* {
    void* p = ws + off;
    off += (bytes + 255) & ~(size_t)255;
    return p;
  };
  __bf16* xb   = (__bf16*)carve((size_t)MROWS * EDIM * 2);
  __bf16* Wqb  = (__bf16*)carve((size_t)EDIM * EDIM * 2);
  __bf16* Wkvb = (__bf16*)carve((size_t)EDIM * KVCOLS * 2);
  __bf16* Wob  = (__bf16*)carve((size_t)EDIM * EDIM * 2);
  __bf16* Qb   = (__bf16*)carve((size_t)MROWS * EDIM * 2);
  __bf16* KVb  = (__bf16*)carve((size_t)MROWS * KVCOLS * 2);
  __bf16* Ab   = (__bf16*)carve((size_t)MROWS * EDIM * 2);

  cvt_f32_bf16<<<2048, 256, 0, stream>>>(x,   xb,   MROWS * EDIM);
  cvt_f32_bf16<<<1024, 256, 0, stream>>>(Wq,  Wqb,  EDIM * EDIM);
  cvt_f32_bf16<<<512,  256, 0, stream>>>(Wkv, Wkvb, EDIM * KVCOLS);
  cvt_f32_bf16<<<1024, 256, 0, stream>>>(Wo,  Wob,  EDIM * EDIM);

  // Q = x @ Wq + bq   (bf16 out)
  gemm_bf16_wmma<<<dim3(EDIM / 64, MROWS / 128), 256, 0, stream>>>(
      xb, Wqb, bq, nullptr, Qb, MROWS, EDIM, EDIM);
  // KV = x @ Wkv + bkv (bf16 out)
  gemm_bf16_wmma<<<dim3(KVCOLS / 64, MROWS / 128), 256, 0, stream>>>(
      xb, Wkvb, bkv, nullptr, KVb, MROWS, KVCOLS, EDIM);
  // attention
  gqa_attention_wmma<<<dim3(SEQ / 128, HEADS, BATCH), 256, 0, stream>>>(
      Qb, KVb, Ab);
  // out = attn @ Wo + bo (fp32 out)
  gemm_bf16_wmma<<<dim3(EDIM / 64, MROWS / 128), 256, 0, stream>>>(
      Ab, Wob, bo, out, nullptr, MROWS, EDIM, EDIM);
}
